// G_enc_29626684407957
// MI455X (gfx1250) — matrix operand order repaired and checked by reading, not compiled, not run on hardware
//
#include <hip/hip_runtime.h>
#include <hip/hip_bf16.h>

// ---------------------------------------------------------------------------
// Problem constants (fixed by the reference setup)
// ---------------------------------------------------------------------------
constexpr int BB = 2048;   // batch
constexpr int TT = 60;     // time
constexpr int DD = 256;    // input dim
constexpr int HH = 128;    // hidden dim
constexpr int KK = 12;     // prediction steps
constexpr int TS = 40;     // t_samples
constexpr int SEQ = TS + 1;
constexpr int RCHUNKS = 16;            // row chunks for 2-stage column argmax
constexpr int RCH = BB / RCHUNKS;      // 128 rows per chunk

typedef __attribute__((ext_vector_type(16))) __bf16 v16bf;
typedef __attribute__((ext_vector_type(8)))  float  v8f;

union BF16x16 {          // 16 bf16 = 32 bytes = one WMMA A/B fragment per lane
    v16bf v;
    uint4 q[2];
};

__device__ __forceinline__ v8f zero8() {
    v8f z = {0.f, 0.f, 0.f, 0.f, 0.f, 0.f, 0.f, 0.f};
    return z;
}

__device__ __forceinline__ v8f wmma_bf16(const BF16x16& a, const BF16x16& b, v8f c) {
    return __builtin_amdgcn_wmma_f32_16x16x32_bf16(false, a.v, false, b.v,
                                                   (short)0, c, false, false);
}

// A fragment: 16x32 bf16, row-major source row `row` (per-lane row = b0 + (lane&15)).
// Lane<16 holds K = k0+{0..7,16..23}; lane>=16 holds K = k0+{8..15,24..31}.
__device__ __forceinline__ void load_fragA(BF16x16& f, const __bf16* row, int k0, int lane) {
    const int kb = k0 + ((lane >> 4) << 3);
    f.q[0] = *(const uint4*)(row + kb);
    f.q[1] = *(const uint4*)(row + kb + 16);
}

// B fragment: 32x16 bf16 where the *source* is row-major [N][Kfull] (weights as
// stored: W[n][k] == Bmat[k][n]).  Lane: n = lane&15, K = k0+{0..7,16..23} (lane<16) / k0+{8..15,24..31} (lane>=16).
__device__ __forceinline__ void load_fragB(BF16x16& f, const __bf16* base,
                                           long rowStride, int n0, int k0, int lane) {
    const int n  = lane & 15;
    const int kb = k0 + ((lane >> 4) << 3);
    const __bf16* r = base + (long)(n0 + n) * rowStride + kb;
    f.q[0] = *(const uint4*)(r);
    f.q[1] = *(const uint4*)(r + 16);
}

__device__ __forceinline__ float sigmoidf_(float x) {
    return 1.0f / (1.0f + __expf(-x));
}

// ---------------------------------------------------------------------------
// Elementwise helpers
// ---------------------------------------------------------------------------
__global__ void cvt_f32_bf16_kernel(const float* __restrict__ src,
                                    __bf16* __restrict__ dst, long n) {
    long i = (long)blockIdx.x * blockDim.x + threadIdx.x;
    long stride = (long)gridDim.x * blockDim.x;
    for (; i < n; i += stride) dst[i] = (__bf16)src[i];
}

__global__ void init_h_kernel(const float* __restrict__ hidden,
                              float* __restrict__ hf0, __bf16* __restrict__ hb0) {
    long i = (long)blockIdx.x * blockDim.x + threadIdx.x;
    if (i < (long)BB * HH) {
        float v = hidden[i];
        hf0[i] = v;
        hb0[i] = (__bf16)v;
    }
}

__global__ void copy_h_kernel(const float* __restrict__ hf, float* __restrict__ out) {
    long i = (long)blockIdx.x * blockDim.x + threadIdx.x;
    if (i < (long)BB * HH) out[2 + i] = hf[i];
}

// ---------------------------------------------------------------------------
// Fused GRU step: gi = x_t@W_ihT, gh = h@W_hhT, gate math, new state (fp32+bf16)
// Block = 128 threads (4 waves). Block owns 16 batch rows; wave w owns h-cols
// [32w, 32w+32). Accumulators: r,z combined (i+h), n split (iN, hN).
// ---------------------------------------------------------------------------
__global__ void __launch_bounds__(128)
gru_step_kernel(const __bf16* __restrict__ zb,
                const __bf16* __restrict__ wihb,   // [3H][D]
                const __bf16* __restrict__ whhb,   // [3H][H]
                const float*  __restrict__ b_ih,
                const float*  __restrict__ b_hh,
                const __bf16* __restrict__ hbIn,   // [B][H]
                const float*  __restrict__ hfIn,   // [B][H]
                __bf16* __restrict__ hbOut,
                float*  __restrict__ hfOut,
                int t) {
    const int lane = threadIdx.x & 31;
    const int wave = threadIdx.x >> 5;
    const int b0   = blockIdx.x * 16;
    const int c0   = wave * 32;

    v8f acc[4][2];   // [0]=r(i+h) [1]=z(i+h) [2]=iN [3]=hN ; 2 col subtiles
#pragma unroll
    for (int g = 0; g < 4; ++g)
#pragma unroll
        for (int s = 0; s < 2; ++s) acc[g][s] = zero8();

    // ---- phase 1: x_t @ W_ih^T   (K = D = 256) ----
    const __bf16* xrow = zb + ((long)(b0 + (lane & 15)) * TT + t) * DD;
#pragma unroll
    for (int k0 = 0; k0 < DD; k0 += 32) {
        BF16x16 a;
        load_fragA(a, xrow, k0, lane);
#pragma unroll
        for (int g = 0; g < 3; ++g) {
#pragma unroll
            for (int s = 0; s < 2; ++s) {
                BF16x16 bq;
                load_fragB(bq, wihb, DD, g * HH + c0 + s * 16, k0, lane);
                const int ai = (g < 2) ? g : 2;
                acc[ai][s] = wmma_bf16(a, bq, acc[ai][s]);
            }
        }
    }

    // ---- phase 2: h @ W_hh^T   (K = H = 128) ----
    const __bf16* hrow = hbIn + (long)(b0 + (lane & 15)) * HH;
#pragma unroll
    for (int k0 = 0; k0 < HH; k0 += 32) {
        BF16x16 a;
        load_fragA(a, hrow, k0, lane);
#pragma unroll
        for (int g = 0; g < 3; ++g) {
#pragma unroll
            for (int s = 0; s < 2; ++s) {
                BF16x16 bq;
                load_fragB(bq, whhb, HH, g * HH + c0 + s * 16, k0, lane);
                const int ai = (g < 2) ? g : 3;
                acc[ai][s] = wmma_bf16(a, bq, acc[ai][s]);
            }
        }
    }

    // ---- gate math + state update ----
    const int n  = lane & 15;
    const int mo = (lane >> 4) << 3;
#pragma unroll
    for (int s = 0; s < 2; ++s) {
        const int col = c0 + s * 16 + n;
        const float bir = b_ih[col],          bhr = b_hh[col];
        const float biz = b_ih[HH + col],     bhz = b_hh[HH + col];
        const float bin = b_ih[2 * HH + col], bhn = b_hh[2 * HH + col];
#pragma unroll
        for (int r = 0; r < 8; ++r) {
            const long row = b0 + r + mo;
            const float rg = sigmoidf_(acc[0][s][r] + bir + bhr);
            const float zg = sigmoidf_(acc[1][s][r] + biz + bhz);
            const float ng = tanhf(acc[2][s][r] + bin + rg * (acc[3][s][r] + bhn));
            const float hold = hfIn[row * HH + col];
            const float hnew = (1.0f - zg) * ng + zg * hold;
            hfOut[row * HH + col] = hnew;
            hbOut[row * HH + col] = (__bf16)hnew;
        }
    }
}

// ---------------------------------------------------------------------------
// pred[k] = c_t @ wk_w[k]^T + wk_b[k]  -> bf16 [K][B][D]
// Grid (B/16, K); block 128 = 4 waves; wave owns 64 d-cols.
// ---------------------------------------------------------------------------
__global__ void __launch_bounds__(128)
pred_kernel(const __bf16* __restrict__ hb,     // c_t bf16 [B][H]
            const __bf16* __restrict__ wkb,    // [K][D][H]
            const float*  __restrict__ wk_b,   // [K][D]
            __bf16* __restrict__ predb) {      // [K][B][D]
    const int lane = threadIdx.x & 31;
    const int wave = threadIdx.x >> 5;
    const int b0   = blockIdx.x * 16;
    const int k    = blockIdx.y;
    const int d0   = wave * 64;

    v8f acc[4];
#pragma unroll
    for (int s = 0; s < 4; ++s) acc[s] = zero8();

    const __bf16* arow  = hb + (long)(b0 + (lane & 15)) * HH;
    const __bf16* wbase = wkb + (long)k * DD * HH;
#pragma unroll
    for (int k0 = 0; k0 < HH; k0 += 32) {
        BF16x16 a;
        load_fragA(a, arow, k0, lane);
#pragma unroll
        for (int s = 0; s < 4; ++s) {
            BF16x16 bq;
            load_fragB(bq, wbase, HH, d0 + s * 16, k0, lane);
            acc[s] = wmma_bf16(a, bq, acc[s]);
        }
    }

    const int n  = lane & 15;
    const int mo = (lane >> 4) << 3;
#pragma unroll
    for (int s = 0; s < 4; ++s) {
        const int d = d0 + s * 16 + n;
        const float bias = wk_b[k * DD + d];
#pragma unroll
        for (int r = 0; r < 8; ++r) {
            const long row = b0 + r + mo;
            predb[((long)k * BB + row) * DD + d] = (__bf16)(acc[s][r] + bias);
        }
    }
}

// ---------------------------------------------------------------------------
// total[k] = enc[k] @ pred[k]^T  -> fp32 buf [B][B]   (one k per launch)
// Grid (B/32, B/256); block 128 = 4 waves; wave owns a 32x64 tile
// (2 A frags x 4 B frags -> 8 WMMA per K-step; 21 FLOP/B from L2).
// ---------------------------------------------------------------------------
__global__ void __launch_bounds__(128)
total_gemm_kernel(const __bf16* __restrict__ zb,
                  const __bf16* __restrict__ predb,
                  float* __restrict__ buf, int k) {
    const int lane = threadIdx.x & 31;
    const int wave = threadIdx.x >> 5;
    const int b0   = blockIdx.x * 32;
    const int c0   = blockIdx.y * 256 + wave * 64;
    const int t    = TS + 1 + k;

    v8f acc[2][4];
#pragma unroll
    for (int i = 0; i < 2; ++i)
#pragma unroll
        for (int s = 0; s < 4; ++s) acc[i][s] = zero8();

    const __bf16* arow0 = zb + ((long)(b0 + (lane & 15)) * TT + t) * DD;
    const __bf16* arow1 = zb + ((long)(b0 + 16 + (lane & 15)) * TT + t) * DD;
    const __bf16* pbase = predb + (long)k * BB * DD;   // rows indexed by column c
#pragma unroll
    for (int k0 = 0; k0 < DD; k0 += 32) {
        BF16x16 a0, a1;
        load_fragA(a0, arow0, k0, lane);
        load_fragA(a1, arow1, k0, lane);
#pragma unroll
        for (int s = 0; s < 4; ++s) {
            BF16x16 bq;
            load_fragB(bq, pbase, DD, c0 + s * 16, k0, lane);
            acc[0][s] = wmma_bf16(a0, bq, acc[0][s]);
            acc[1][s] = wmma_bf16(a1, bq, acc[1][s]);
        }
    }

    const int n  = lane & 15;
    const int mo = (lane >> 4) << 3;
#pragma unroll
    for (int i = 0; i < 2; ++i) {
#pragma unroll
        for (int s = 0; s < 4; ++s) {
            const int c = c0 + s * 16 + n;
#pragma unroll
            for (int r = 0; r < 8; ++r) {
                const long row = b0 + i * 16 + r + mo;
                buf[row * BB + c] = acc[i][s][r];
            }
        }
    }
}

// ---------------------------------------------------------------------------
// Per-row online logsumexp (single pass over buf), diagonal log-softmax entry.
// Deterministic tree combine of (max, scaled-sum) pairs.
// ---------------------------------------------------------------------------
__global__ void __launch_bounds__(256)
row_lse_kernel(const float* __restrict__ buf, float* __restrict__ lsdiag, int k) {
    __shared__ float sm_m[256];
    __shared__ float sm_s[256];
    const int b   = blockIdx.x;
    const int tid = threadIdx.x;
    const float* row = buf + (long)b * BB;

    float mx = -INFINITY, sum = 0.f;
    for (int i = tid; i < BB; i += 256) {
        const float v = row[i];
        if (v > mx) {
            sum = sum * __expf(mx - v) + 1.0f;
            mx = v;
        } else {
            sum += __expf(v - mx);
        }
    }
    sm_m[tid] = mx;
    sm_s[tid] = sum;
    __syncthreads();
    for (int st = 128; st > 0; st >>= 1) {
        if (tid < st) {
            const float m1 = sm_m[tid], s1 = sm_s[tid];
            const float m2 = sm_m[tid + st], s2 = sm_s[tid + st];
            const float M = fmaxf(m1, m2);
            sm_m[tid] = M;
            sm_s[tid] = s1 * __expf(m1 - M) + s2 * __expf(m2 - M);
        }
        __syncthreads();
    }
    if (tid == 0) lsdiag[(long)k * BB + b] = row[b] - (sm_m[0] + logf(sm_s[0]));
}

// ---------------------------------------------------------------------------
// Column argmax, stage 1 (coalesced): thread = column, block = (col-chunk,
// row-chunk). Consecutive lanes read consecutive addresses every iteration.
// ---------------------------------------------------------------------------
__global__ void __launch_bounds__(256)
col_argmax_part_kernel(const float* __restrict__ buf,
                       float* __restrict__ pv, int* __restrict__ pi) {
    const int c  = blockIdx.x * 256 + threadIdx.x;
    const int r0 = blockIdx.y * RCH;

    float bv = -INFINITY;
    int   bi = r0;
    for (int r = r0; r < r0 + RCH; ++r) {
        const float v = buf[(long)r * BB + c];
        if (v > bv) { bv = v; bi = r; }        // strictly-greater keeps first max
    }
    pv[(long)blockIdx.y * BB + c] = bv;
    pi[(long)blockIdx.y * BB + c] = bi;
}

// Stage 2: combine the RCHUNKS partials per column in ascending row order
// (strictly-greater replacement preserves jnp.argmax first-max semantics).
__global__ void __launch_bounds__(256)
col_argmax_comb_kernel(const float* __restrict__ pv, const int* __restrict__ pi,
                       float* __restrict__ colhit, int k) {
    const int c = blockIdx.x * 256 + threadIdx.x;
    float bv = pv[c];
    int   bi = pi[c];
#pragma unroll
    for (int j = 1; j < RCHUNKS; ++j) {
        const float v = pv[(long)j * BB + c];
        if (v > bv) { bv = v; bi = pi[(long)j * BB + c]; }
    }
    colhit[(long)k * BB + c] = (bi == c) ? 1.0f : 0.0f;
}

// ---------------------------------------------------------------------------
// Final deterministic reductions -> accuracy, nce, accuracies[K]
// out layout: [accuracy, nce, h_last(B*H), accuracies(K)]
// ---------------------------------------------------------------------------
__global__ void __launch_bounds__(256)
finalize_kernel(const float* __restrict__ lsdiag,
                const float* __restrict__ colhit,
                float* __restrict__ out) {
    __shared__ float sm[256];
    const int tid = threadIdx.x;

    float s = 0.f;
    for (int i = tid; i < KK * BB; i += 256) s += lsdiag[i];
    sm[tid] = s;
    __syncthreads();
    for (int st = 128; st > 0; st >>= 1) {
        if (tid < st) sm[tid] += sm[tid + st];
        __syncthreads();
    }
    if (tid == 0) out[1] = sm[0] / (-1.0f * BB * KK);
    __syncthreads();

    for (int k = 0; k < KK; ++k) {
        float a = 0.f;
        for (int i = tid; i < BB; i += 256) a += colhit[(long)k * BB + i];
        sm[tid] = a;
        __syncthreads();
        for (int st = 128; st > 0; st >>= 1) {
            if (tid < st) sm[tid] += sm[tid + st];
            __syncthreads();
        }
        if (tid == 0) {
            const float acc = sm[0] / (float)BB;
            out[2 + (long)BB * HH + k] = acc;
            if (k == KK - 1) out[0] = acc;
        }
        __syncthreads();
    }
}

// ---------------------------------------------------------------------------
// Host-side orchestration (stream-ordered only; graph-capture safe)
// ---------------------------------------------------------------------------
extern "C" void kernel_launch(void* const* d_in, const int* in_sizes, int n_in,
                              void* d_out, int out_size, void* d_ws, size_t ws_size,
                              hipStream_t stream) {
    (void)in_sizes; (void)n_in; (void)out_size; (void)ws_size;

    const float* z      = (const float*)d_in[0];
    const float* hidden = (const float*)d_in[1];
    const float* w_ih   = (const float*)d_in[2];
    const float* w_hh   = (const float*)d_in[3];
    const float* b_ih   = (const float*)d_in[4];
    const float* b_hh   = (const float*)d_in[5];
    const float* wk_w   = (const float*)d_in[6];
    const float* wk_b   = (const float*)d_in[7];
    float* out = (float*)d_out;

    // ---- workspace carve-up (256B aligned) ----
    char* ws = (char*)d_ws;
    size_t off = 0;
    auto carve = [&](size_t bytes) {
        char* p = ws + off;
        off += (bytes + 255) & ~size_t(255);
        return p;
    };
    __bf16* zb    = (__bf16*)carve((size_t)BB * TT * DD * 2);   // 62.9 MB
    __bf16* wihb  = (__bf16*)carve((size_t)3 * HH * DD * 2);
    __bf16* whhb  = (__bf16*)carve((size_t)3 * HH * HH * 2);
    __bf16* wkb   = (__bf16*)carve((size_t)KK * DD * HH * 2);
    __bf16* hb0   = (__bf16*)carve((size_t)BB * HH * 2);
    __bf16* hb1   = (__bf16*)carve((size_t)BB * HH * 2);
    float*  hf0   = (float*) carve((size_t)BB * HH * 4);
    float*  hf1   = (float*) carve((size_t)BB * HH * 4);
    __bf16* predb = (__bf16*)carve((size_t)KK * BB * DD * 2);   // 12.6 MB
    float*  buf   = (float*) carve((size_t)BB * BB * 4);        // 16.8 MB (reused per k)
    float*  lsdg  = (float*) carve((size_t)KK * BB * 4);
    float*  chit  = (float*) carve((size_t)KK * BB * 4);
    float*  pv    = (float*) carve((size_t)RCHUNKS * BB * 4);   // argmax partial vals
    int*    pi    = (int*)   carve((size_t)RCHUNKS * BB * 4);   // argmax partial idx

    // ---- one-time bf16 conversions ----
    cvt_f32_bf16_kernel<<<4096, 256, 0, stream>>>(z, zb, (long)BB * TT * DD);
    cvt_f32_bf16_kernel<<<96,   256, 0, stream>>>(w_ih, wihb, (long)3 * HH * DD);
    cvt_f32_bf16_kernel<<<96,   256, 0, stream>>>(w_hh, whhb, (long)3 * HH * HH);
    cvt_f32_bf16_kernel<<<384,  256, 0, stream>>>(wk_w, wkb, (long)KK * DD * HH);
    init_h_kernel<<<(BB * HH + 255) / 256, 256, 0, stream>>>(hidden, hf0, hb0);

    // ---- 41 fused GRU steps (ping-pong state) ----
    for (int t = 0; t < SEQ; ++t) {
        const bool even = (t & 1) == 0;
        const __bf16* hbi = even ? hb0 : hb1;
        const float*  hfi = even ? hf0 : hf1;
        __bf16* hbo = even ? hb1 : hb0;
        float*  hfo = even ? hf1 : hf0;
        gru_step_kernel<<<BB / 16, 128, 0, stream>>>(zb, wihb, whhb, b_ih, b_hh,
                                                     hbi, hfi, hbo, hfo, t);
    }
    // SEQ=41 steps, last step t=40 (even) wrote hb1/hf1 -> c_t lives there.
    const __bf16* ctb = hb1;
    const float*  ctf = hf1;

    // ---- pred[k] = c_t @ wk_w[k]^T + wk_b[k] ----
    pred_kernel<<<dim3(BB / 16, KK), 128, 0, stream>>>(ctb, wkb, wk_b, predb);

    // ---- per-k: total GEMM + reductions (staged in reused 16 MB buffer) ----
    for (int k = 0; k < KK; ++k) {
        total_gemm_kernel<<<dim3(BB / 32, BB / 256), 128, 0, stream>>>(zb, predb, buf, k);
        row_lse_kernel<<<BB, 256, 0, stream>>>(buf, lsdg, k);
        col_argmax_part_kernel<<<dim3(BB / 256, RCHUNKS), 256, 0, stream>>>(buf, pv, pi);
        col_argmax_comb_kernel<<<BB / 256, 256, 0, stream>>>(pv, pi, chit, k);
    }

    // ---- outputs ----
    finalize_kernel<<<1, 256, 0, stream>>>(lsdg, chit, out);
    copy_h_kernel<<<(BB * HH + 255) / 256, 256, 0, stream>>>(ctf, out);
}
